// SelfAttention_85169201480424
// MI455X (gfx1250) — matrix operand order repaired and checked by reading, not compiled, not run on hardware
//
#include <hip/hip_runtime.h>
#include <math.h>

// ---------------------------------------------------------------------------
// Self-attention forward for MI455X (gfx1250, wave32, WMMA, async-to-LDS).
// B=2, N=4096, D=512, H=8, HD=64.  All matmuls bf16 WMMA w/ f32 accum.
// ---------------------------------------------------------------------------

typedef __bf16 bf16;
typedef __attribute__((ext_vector_type(16))) __bf16 bf16x16;
typedef __attribute__((ext_vector_type(8)))  __bf16 bf16x8;
typedef __attribute__((ext_vector_type(4)))  __bf16 bf16x4;
typedef __attribute__((ext_vector_type(8)))  float  f32x8;
typedef __attribute__((ext_vector_type(4)))  float  f32x4;

#define WMMA_BF16(a, b, c) \
  __builtin_amdgcn_wmma_f32_16x16x32_bf16(false, (a), false, (b), (short)0, (c), false, false)

static constexpr int Bsz = 2, Nseq = 4096, Dm = 512, Hh = 8, Hd = 64;
static constexpr int BH = Bsz * Hh;                          // 16
static constexpr size_t HEAD_ELEMS = (size_t)BH * Nseq * Hd; // 4,194,304

// ---------------------------------------------------------------------------
// CDNA5 async global->LDS copy (ASYNCcnt path, CDNA5 ISA).
// ---------------------------------------------------------------------------
__device__ __forceinline__ void async_b128(unsigned lds_byte, const void* g) {
  asm volatile("global_load_async_to_lds_b128 %0, %1, off"
               :: "v"(lds_byte), "v"(g) : "memory");
}
__device__ __forceinline__ void wait_async0() {
  asm volatile("s_wait_asynccnt 0x0" ::: "memory");
}
__device__ __forceinline__ unsigned lds_off(const void* p) {
  return (unsigned)(unsigned long long)p;
}
// Pin an LDS *element offset* after the sync point: dependent ds_loads cannot
// be hoisted above s_wait_asynccnt / the barrier, while the __shared__ base
// keeps address-space inference intact (loads stay on the DS path, not FLAT).
__device__ __forceinline__ int pin_i(int v) {
  asm volatile("" : "+v"(v));
  return v;
}

// ---------------------------------------------------------------------------
// Fragment loaders per CDNA5 WMMA VGPR layouts (CDNA5 ISA).
// ---------------------------------------------------------------------------
__device__ __forceinline__ bf16x16 load_a_frag(const bf16* base, int stride,
                                               int lane, int k0) {
  int row = lane & 15;
  int kh  = (lane >> 4) & 1;
  const bf16* p = base + row * stride + k0 + kh * 8;
  bf16x8 lo = *(const bf16x8*)(p);
  bf16x8 hi = *(const bf16x8*)(p + 16);
  bf16x16 r;
#pragma unroll
  for (int i = 0; i < 8; ++i) { r[i] = lo[i]; r[i + 8] = hi[i]; }
  return r;
}

__device__ __forceinline__ bf16x16 load_b_frag(const bf16* base, int stride,
                                               int lane, int k0) {
  int n = lane & 15;
  const bf16* p = base + n * stride + k0 + ((lane >> 4) << 3);
  bf16x8 lo = *(const bf16x8*)(p);
  bf16x8 hi = *(const bf16x8*)(p + 16);
  bf16x16 r;
#pragma unroll
  for (int i = 0; i < 8; ++i) { r[i] = lo[i]; r[i + 8] = hi[i]; }
  return r;
}

// ---------------------------------------------------------------------------
// Prep kernel A: x f32 -> bf16 (one pass, so GEMM tiles become byte copies).
// ---------------------------------------------------------------------------
__global__ __launch_bounds__(256) void cvt_kernel(const float* __restrict__ src,
                                                  bf16* __restrict__ dst, int n4) {
  int i = blockIdx.x * 256 + threadIdx.x;
  if (i < n4) {
    f32x4 v = *(const f32x4*)(src + (size_t)i * 4);
    bf16x4 o;
#pragma unroll
    for (int j = 0; j < 4; ++j) o[j] = (bf16)v[j];
    *(bf16x4*)(dst + (size_t)i * 4) = o;
  }
}

// ---------------------------------------------------------------------------
// Prep kernel B: W[k][n] f32 -> Wt[n][k] bf16 via LDS-tiled transpose.
// grid (16,16,4): 32x32 tiles, z selects the matrix.
// ---------------------------------------------------------------------------
__global__ __launch_bounds__(256) void wtrans_kernel(
    const float* __restrict__ Wq, const float* __restrict__ Wk,
    const float* __restrict__ Wv, const float* __restrict__ Wo,
    bf16* __restrict__ Wqt, bf16* __restrict__ Wkt,
    bf16* __restrict__ Wvt, bf16* __restrict__ Wot) {
  const int zi = blockIdx.z;
  const float* W = (zi == 0) ? Wq : (zi == 1) ? Wk : (zi == 2) ? Wv : Wo;
  bf16* Wt       = (zi == 0) ? Wqt : (zi == 1) ? Wkt : (zi == 2) ? Wvt : Wot;
  __shared__ bf16 tile[32][33];
  const int bn = blockIdx.x * 32, bk = blockIdx.y * 32;
  const int tx = threadIdx.x & 31, ty = threadIdx.x >> 5;  // 32x8
#pragma unroll
  for (int r = 0; r < 32; r += 8)
    tile[ty + r][tx] = (bf16)W[(size_t)(bk + ty + r) * Dm + bn + tx];
  __syncthreads();
#pragma unroll
  for (int r = 0; r < 32; r += 8)
    Wt[(size_t)(bn + ty + r) * Dm + bk + tx] = tile[tx][ty + r];
}

// ---------------------------------------------------------------------------
// Kernel 1: QKV projection GEMM, async double-buffered LDS staging.
// xb[8192,512] bf16 @ Wt[512(n),512(k)] bf16 + bias -> [B,H,N,HD] bf16.
// Block tile 128x128, K-step 32; 8 waves x 32x64 (2x4 WMMA).  grid (4,64,3).
// ---------------------------------------------------------------------------
__global__ __launch_bounds__(256) void qkv_kernel(
    const bf16* __restrict__ xb,
    const bf16* __restrict__ Wqt, const float* __restrict__ bq,
    const bf16* __restrict__ Wkt, const float* __restrict__ bk,
    const bf16* __restrict__ Wvt, const float* __restrict__ bv,
    bf16* __restrict__ Qo, bf16* __restrict__ Ko, bf16* __restrict__ Vo) {
  constexpr int BM = 128, BN = 128, BK = 32, ASTR = 40, BSTR = 40, NT = Dm / BK;
  constexpr int ABUF = BM * ASTR, BBUF = BN * BSTR;
  __shared__ bf16 As[2 * ABUF];
  __shared__ bf16 Bs[2 * BBUF];

  const int zi = blockIdx.z;
  const bf16* Wt   = (zi == 0) ? Wqt : ((zi == 1) ? Wkt : Wvt);
  const float* bia = (zi == 0) ? bq : ((zi == 1) ? bk : bv);
  bf16* Out        = (zi == 0) ? Qo : ((zi == 1) ? Ko : Vo);

  const int tid = threadIdx.x, lane = tid & 31, wave = tid >> 5;
  const int waveM = wave >> 1, waveN = wave & 1;
  const int mBlock = blockIdx.y * BM, nBlock = blockIdx.x * BN;

  // Per-thread copy assignments (pure byte copies -> async engine).
  const int arow = tid >> 1, ac0 = (tid & 1) * 16;  // A: 2x b128 / thread
  const int brow = tid >> 1, bc0 = (tid & 1) * 16;  // B: 2x b128 / thread

  auto issueAB = [&](int kt, int buf) {
    const bf16* ga = xb + (size_t)(mBlock + arow) * Dm + kt + ac0;
    unsigned la = lds_off(&As[buf * ABUF + arow * ASTR + ac0]);
    async_b128(la, ga);
    async_b128(la + 16, ga + 8);
    const bf16* gb = Wt + (size_t)(nBlock + brow) * Dm + kt + bc0;
    unsigned lb = lds_off(&Bs[buf * BBUF + brow * BSTR + bc0]);
    async_b128(lb, gb);
    async_b128(lb + 16, gb + 8);
  };

  f32x8 acc[2][4] = {};
  issueAB(0, 0);
  for (int t = 0; t < NT; ++t) {
    const int cur = t & 1;
    wait_async0();
    __syncthreads();
    const int aoff = pin_i(cur * ABUF + (waveM * 32) * ASTR);
    const int boff = pin_i(cur * BBUF + (waveN * 64) * BSTR);
    if (t + 1 < NT) issueAB((t + 1) * BK, 1 - cur);

    bf16x16 a0 = load_a_frag(&As[aoff],             ASTR, lane, 0);
    bf16x16 a1 = load_a_frag(&As[aoff + 16 * ASTR], ASTR, lane, 0);
#pragma unroll
    for (int j = 0; j < 4; ++j) {
      bf16x16 b = load_b_frag(&Bs[boff + (j * 16) * BSTR], BSTR, lane, 0);
      acc[0][j] = WMMA_BF16(a0, b, acc[0][j]);
      acc[1][j] = WMMA_BF16(a1, b, acc[1][j]);
    }
  }

  // Epilogue: +bias, split heads -> [B,H,N,HD] bf16.
#pragma unroll
  for (int i2 = 0; i2 < 2; ++i2)
#pragma unroll
    for (int j2 = 0; j2 < 4; ++j2) {
      int col = nBlock + waveN * 64 + j2 * 16 + (lane & 15);
      float bvv = bia[col];
      int h = col >> 6, hd = col & 63;
#pragma unroll
      for (int i = 0; i < 8; ++i) {
        int gr = mBlock + waveM * 32 + i2 * 16 + i + ((lane >> 4) * 8);
        int bidx = gr >> 12, n = gr & (Nseq - 1);
        float v = acc[i2][j2][i] + bvv;
        Out[(((size_t)bidx * Hh + h) * Nseq + n) * Hd + hd] = (bf16)v;
      }
    }
}

// ---------------------------------------------------------------------------
// Kernel 2: flash attention per (b,h); 128 query rows per workgroup
// (8 waves x 16 rows), 64-key tiles, async double-buffered K, manual
// transposed V (double-buffered in the same slot).  grid (32, 16).
// ---------------------------------------------------------------------------
__global__ __launch_bounds__(256) void attn_kernel(
    const bf16* __restrict__ Qg, const bf16* __restrict__ Kg,
    const bf16* __restrict__ Vg, bf16* __restrict__ Cg) {
  constexpr int QB = 128, KB = 64, NT = Nseq / KB;
  constexpr int QSTR = 72, KSTR = 72, VSTR = 72, PSTR = 72;
  constexpr int KBUF = KB * KSTR, VBUF = Hd * VSTR;
  __shared__ bf16 Qs[QB * QSTR];
  __shared__ bf16 Ks[2 * KBUF];
  __shared__ bf16 Vs[2 * VBUF];         // transposed: [d][key]
  __shared__ bf16 Ps[8 * 16 * PSTR];    // per-wave P staging [wave][16][64]

  const int tid = threadIdx.x, lane = tid & 31, wave = tid >> 5;
  const int bh = blockIdx.y, b = bh >> 3, h = bh & 7;
  const int qBase = blockIdx.x * QB;
  const bf16* Qh = Qg + (size_t)bh * Nseq * Hd;
  const bf16* Kh = Kg + (size_t)bh * Nseq * Hd;
  const bf16* Vh = Vg + (size_t)bh * Nseq * Hd;

  const int krow = tid >> 2, kc0 = (tid & 3) * 16;  // K copy: 2x b128/thread
  auto issueK = [&](int k0, int buf) {
    const bf16* g = Kh + (size_t)(k0 + krow) * Hd + kc0;
    unsigned l = lds_off(&Ks[buf * KBUF + krow * KSTR + kc0]);
    async_b128(l, g);
    async_b128(l + 16, g + 8);
  };
  auto copyV = [&](int k0, int buf) {  // transpose: Vs[d][key]
    const bf16* s = Vh + (size_t)(k0 + krow) * Hd + kc0;
    bf16x8 v0 = *(const bf16x8*)(s + 0);
    bf16x8 v1 = *(const bf16x8*)(s + 8);
    const int vb = buf * VBUF;
#pragma unroll
    for (int j = 0; j < 8; ++j) {
      Vs[vb + (kc0 + j) * VSTR + krow]     = v0[j];
      Vs[vb + (kc0 + 8 + j) * VSTR + krow] = v1[j];
    }
  };

  {  // stage Q block 128x64 via async (4x b128/thread)
    int row = tid >> 1, c0 = (tid & 1) * 32;
    const bf16* g = Qh + (size_t)(qBase + row) * Hd + c0;
    unsigned l = lds_off(&Qs[row * QSTR + c0]);
    async_b128(l, g);
    async_b128(l + 16, g + 8);
    async_b128(l + 32, g + 16);
    async_b128(l + 48, g + 24);
  }
  issueK(0, 0);
  copyV(0, 0);
  wait_async0();
  __syncthreads();

  bf16x16 qa0, qa1;
  {
    const int qoff = pin_i(wave * 16 * QSTR);
    qa0 = load_a_frag(&Qs[qoff], QSTR, lane, 0);
    qa1 = load_a_frag(&Qs[qoff], QSTR, lane, 32);
  }

  f32x8 acc[4] = {};
  float m_i[8], l_i[8];
#pragma unroll
  for (int i = 0; i < 8; ++i) { m_i[i] = -INFINITY; l_i[i] = 0.0f; }
  const float scale = 0.044194173824159216f;  // 1/sqrt(D=512) per reference

  for (int t = 0; t < NT; ++t) {
    const int cur = t & 1;
    if (t) { wait_async0(); __syncthreads(); }
    const int koff = pin_i(cur * KBUF);
    const int voff = pin_i(cur * VBUF);
    if (t + 1 < NT) { issueK((t + 1) * KB, 1 - cur); copyV((t + 1) * KB, 1 - cur); }

    // S = scale * Q K^T : 16x64 per wave (4 column tiles x 2 WMMAs)
    float st[4][8];
#pragma unroll
    for (int c = 0; c < 4; ++c) {
      f32x8 s = {};
      bf16x16 kb0 = load_b_frag(&Ks[koff + c * 16 * KSTR], KSTR, lane, 0);
      bf16x16 kb1 = load_b_frag(&Ks[koff + c * 16 * KSTR], KSTR, lane, 32);
      s = WMMA_BF16(qa0, kb0, s);
      s = WMMA_BF16(qa1, kb1, s);
#pragma unroll
      for (int i = 0; i < 8; ++i) st[c][i] = s[i] * scale;
    }

    // Online softmax (D-layout: VGPR i <-> row i+8*(lane>=16), col=lane&15).
    const int pbase = wave * 16 * PSTR;
    const int prow_hi = (lane >> 4) * 8;
#pragma unroll
    for (int i = 0; i < 8; ++i) {
      float v = fmaxf(fmaxf(st[0][i], st[1][i]), fmaxf(st[2][i], st[3][i]));
      v = fmaxf(v, __shfl_xor(v, 1, 32));
      v = fmaxf(v, __shfl_xor(v, 2, 32));
      v = fmaxf(v, __shfl_xor(v, 4, 32));
      v = fmaxf(v, __shfl_xor(v, 8, 32));
      float mn   = fmaxf(m_i[i], v);
      float corr = __expf(m_i[i] - mn);
      m_i[i] = mn;
      float sum = 0.0f;
#pragma unroll
      for (int c = 0; c < 4; ++c) {
        float p = __expf(st[c][i] - mn);
        sum += p;
        Ps[pbase + (i + prow_hi) * PSTR + c * 16 + (lane & 15)] = (bf16)p;
      }
      sum += __shfl_xor(sum, 1, 32);
      sum += __shfl_xor(sum, 2, 32);
      sum += __shfl_xor(sum, 4, 32);
      sum += __shfl_xor(sum, 8, 32);
      l_i[i] = l_i[i] * corr + sum;
#pragma unroll
      for (int dg = 0; dg < 4; ++dg) acc[dg][i] *= corr;
    }
    // Intra-wave LDS write->read (same-wave LDS ops complete in order);
    // barrier blocks compiler reordering.
    __builtin_amdgcn_wave_barrier();

    const int poff = pin_i(pbase);
    bf16x16 pa0 = load_a_frag(&Ps[poff], PSTR, lane, 0);
    bf16x16 pa1 = load_a_frag(&Ps[poff], PSTR, lane, 32);
#pragma unroll
    for (int dg = 0; dg < 4; ++dg) {
      bf16x16 vb0 = load_b_frag(&Vs[voff + dg * 16 * VSTR], VSTR, lane, 0);
      bf16x16 vb1 = load_b_frag(&Vs[voff + dg * 16 * VSTR], VSTR, lane, 32);
      acc[dg] = WMMA_BF16(pa0, vb0, acc[dg]);
      acc[dg] = WMMA_BF16(pa1, vb1, acc[dg]);
    }
  }

  // Normalize, write ctx (merged heads) bf16 [B,N,D].
#pragma unroll
  for (int i = 0; i < 8; ++i) {
    float inv = 1.0f / l_i[i];
    int n = qBase + wave * 16 + i + ((lane >> 4) * 8);
    bf16* dst = Cg + ((size_t)b * Nseq + n) * Dm + h * Hd + (lane & 15);
#pragma unroll
    for (int dg = 0; dg < 4; ++dg) dst[dg * 16] = (bf16)(acc[dg][i] * inv);
  }
}

// ---------------------------------------------------------------------------
// Kernel 3: output projection.  ctx bf16 @ Wot bf16 + bo -> f32 out.
// Same async double-buffered 128x128 GEMM skeleton.  grid (4, 64).
// ---------------------------------------------------------------------------
__global__ __launch_bounds__(256) void outproj_kernel(
    const bf16* __restrict__ ctx, const bf16* __restrict__ Wot,
    const float* __restrict__ bo, float* __restrict__ out) {
  constexpr int BM = 128, BN = 128, BK = 32, ASTR = 40, BSTR = 40, NT = Dm / BK;
  constexpr int ABUF = BM * ASTR, BBUF = BN * BSTR;
  __shared__ bf16 As[2 * ABUF];
  __shared__ bf16 Bs[2 * BBUF];

  const int tid = threadIdx.x, lane = tid & 31, wave = tid >> 5;
  const int waveM = wave >> 1, waveN = wave & 1;
  const int mBlock = blockIdx.y * BM, nBlock = blockIdx.x * BN;

  const int arow = tid >> 1, ac0 = (tid & 1) * 16;
  const int brow = tid >> 1, bc0 = (tid & 1) * 16;

  auto issueAB = [&](int kt, int buf) {
    const bf16* ga = ctx + (size_t)(mBlock + arow) * Dm + kt + ac0;
    unsigned la = lds_off(&As[buf * ABUF + arow * ASTR + ac0]);
    async_b128(la, ga);
    async_b128(la + 16, ga + 8);
    const bf16* gb = Wot + (size_t)(nBlock + brow) * Dm + kt + bc0;
    unsigned lb = lds_off(&Bs[buf * BBUF + brow * BSTR + bc0]);
    async_b128(lb, gb);
    async_b128(lb + 16, gb + 8);
  };

  f32x8 acc[2][4] = {};
  issueAB(0, 0);
  for (int t = 0; t < NT; ++t) {
    const int cur = t & 1;
    wait_async0();
    __syncthreads();
    const int aoff = pin_i(cur * ABUF + (waveM * 32) * ASTR);
    const int boff = pin_i(cur * BBUF + (waveN * 64) * BSTR);
    if (t + 1 < NT) issueAB((t + 1) * BK, 1 - cur);

    bf16x16 a0 = load_a_frag(&As[aoff],             ASTR, lane, 0);
    bf16x16 a1 = load_a_frag(&As[aoff + 16 * ASTR], ASTR, lane, 0);
#pragma unroll
    for (int j = 0; j < 4; ++j) {
      bf16x16 b = load_b_frag(&Bs[boff + (j * 16) * BSTR], BSTR, lane, 0);
      acc[0][j] = WMMA_BF16(a0, b, acc[0][j]);
      acc[1][j] = WMMA_BF16(a1, b, acc[1][j]);
    }
  }

#pragma unroll
  for (int i2 = 0; i2 < 2; ++i2)
#pragma unroll
    for (int j2 = 0; j2 < 4; ++j2) {
      int col = nBlock + waveN * 64 + j2 * 16 + (lane & 15);
      float bvv = bo[col];
#pragma unroll
      for (int i = 0; i < 8; ++i) {
        int gr = mBlock + waveM * 32 + i2 * 16 + i + ((lane >> 4) * 8);
        out[(size_t)gr * Dm + col] = acc[i2][j2][i] + bvv;
      }
    }
}

// ---------------------------------------------------------------------------
extern "C" void kernel_launch(void* const* d_in, const int* in_sizes, int n_in,
                              void* d_out, int out_size, void* d_ws,
                              size_t ws_size, hipStream_t stream) {
  (void)in_sizes; (void)n_in; (void)out_size; (void)ws_size;
  const float* x  = (const float*)d_in[0];
  const float* Wq = (const float*)d_in[1];
  const float* bq = (const float*)d_in[2];
  const float* Wk = (const float*)d_in[3];
  const float* bk = (const float*)d_in[4];
  const float* Wv = (const float*)d_in[5];
  const float* bv = (const float*)d_in[6];
  const float* Wo = (const float*)d_in[7];
  const float* bo = (const float*)d_in[8];
  float* out = (float*)d_out;

  // Workspace (bf16): xb | Wqt Wkt Wvt Wot | Q K V | ctx   (~44 MB)
  bf16* xb  = (bf16*)d_ws;
  bf16* Wqt = xb + (size_t)Bsz * Nseq * Dm;
  bf16* Wkt = Wqt + (size_t)Dm * Dm;
  bf16* Wvt = Wkt + (size_t)Dm * Dm;
  bf16* Wot = Wvt + (size_t)Dm * Dm;
  bf16* Qw  = Wot + (size_t)Dm * Dm;
  bf16* Kw  = Qw + HEAD_ELEMS;
  bf16* Vw  = Kw + HEAD_ELEMS;
  bf16* Cw  = Vw + HEAD_ELEMS;

  const int n4 = (Bsz * Nseq * Dm) / 4;
  dim3 blk(256);
  cvt_kernel<<<dim3((n4 + 255) / 256), blk, 0, stream>>>(x, xb, n4);
  wtrans_kernel<<<dim3(16, 16, 4), blk, 0, stream>>>(Wq, Wk, Wv, Wo,
                                                     Wqt, Wkt, Wvt, Wot);
  qkv_kernel<<<dim3(Dm / 128, (Bsz * Nseq) / 128, 3), blk, 0, stream>>>(
      xb, Wqt, bq, Wkt, bk, Wvt, bv, Qw, Kw, Vw);
  attn_kernel<<<dim3(Nseq / 128, BH), blk, 0, stream>>>(Qw, Kw, Vw, Cw);
  outproj_kernel<<<dim3(Dm / 128, (Bsz * Nseq) / 128), blk, 0, stream>>>(
      Cw, Wot, bo, out);
}
